// GATConv_59528246723308
// MI455X (gfx1250) — matrix operand order repaired and checked by reading, not compiled, not run on hardware
//
#include <hip/hip_runtime.h>
#include <hip/hip_bf16.h>
#include <math.h>

// ---- problem constants (match reference) ----
#define IN_FEATS  256
#define NUM_HEADS 4
#define OUT_FEATS 64
#define HD        (NUM_HEADS * OUT_FEATS)   // 256
#define GAT_ALPHA 0.2f
#define BSTRIDE   264   // bf16 units per LDS row: 16B-aligned rows + bank spread

typedef __attribute__((ext_vector_type(16))) __bf16 v16bf;
typedef __attribute__((ext_vector_type(8)))  __bf16 v8bf;
typedef __attribute__((ext_vector_type(4)))  __bf16 v4bf;
typedef __attribute__((ext_vector_type(8)))  float  v8f;

// fp32 -> bf16 round-to-nearest-even via bit manipulation
__device__ __forceinline__ __bf16 f2bf(float f) {
    unsigned u = __float_as_uint(f);
    u += 0x7FFFu + ((u >> 16) & 1u);
    unsigned short h = (unsigned short)(u >> 16);
    __bf16 r;
    __builtin_memcpy(&r, &h, sizeof(r));
    return r;
}

// float atomic max via monotonic int mapping (valid with -inf init)
__device__ __forceinline__ void atomicMaxF(float* addr, float v) {
    if (v >= 0.0f) atomicMax((int*)addr, __float_as_int(v));
    else           atomicMin((unsigned int*)addr, __float_as_uint(v));
}

// ---------------- fill ----------------
__global__ void fill_kernel(float* __restrict__ p, long long n, float v) {
    long long i      = (long long)blockIdx.x * blockDim.x + threadIdx.x;
    long long stride = (long long)gridDim.x * blockDim.x;
    for (; i < n; i += stride) p[i] = v;
}

// ---- convert + transpose weight: Wt[n][k] = bf16(fc_w[k][n]) ----
__global__ void wconv_kernel(const float* __restrict__ w, __bf16* __restrict__ wt) {
    int i = blockIdx.x * 256 + threadIdx.x;   // flat index into Wt [HD][IN_FEATS]
    int n = i >> 8;
    int k = i & 255;
    wt[i] = f2bf(w[k * HD + n]);
}

// ---- WMMA GEMM: ft[16-row strip] = feat_strip @ W  (bf16 inputs, f32 accum) ----
__global__ void __launch_bounds__(256) gemm_kernel(const float* __restrict__ feat,
                                                   const __bf16* __restrict__ wt,
                                                   float* __restrict__ ft, int nNodes) {
    __shared__ __bf16 Ab[16 * BSTRIDE];   // 8.25 KB, bf16 staged once
    const int row0 = blockIdx.x * 16;

    // stage + convert: 16x256 floats -> bf16 LDS, float4 granularity (1024 quads)
    for (int q = threadIdx.x; q < 16 * (IN_FEATS / 4); q += 256) {
        const int r = q >> 6;            // 64 quads per row
        const int c = (q & 63) << 2;
        float4 f;
        f.x = f.y = f.z = f.w = 0.0f;
        const int gr = row0 + r;
        if (gr < nNodes) f = *(const float4*)(feat + (long long)gr * IN_FEATS + c);
        v4bf b;
        b[0] = f2bf(f.x); b[1] = f2bf(f.y); b[2] = f2bf(f.z); b[3] = f2bf(f.w);
        *(v4bf*)&Ab[r * BSTRIDE + c] = b;
    }
    __syncthreads();

    const int wave  = threadIdx.x >> 5;
    const int lane  = threadIdx.x & 31;
    const int M     = lane & 15;        // A row (ISA A-fragment layout)
    const int khalf = lane >> 4;        // A K-half selector
    const int nsel  = lane & 15;        // B column within tile
    const int kb    = (lane >> 4) * 16; // B K-base selector
    const int n0    = wave * 32;        // this wave's two N-tiles: n0, n0+16

    v8f acc0 = {};
    v8f acc1 = {};

    for (int k0 = 0; k0 < IN_FEATS; k0 += 32) {
        // A fragment: element e -> K = k0 + khalf*16 + e (contiguous, matches B)
        // = two contiguous 8-element runs -> two ds b128 loads, no conversion
        const v8bf lo = *(const v8bf*)&Ab[M * BSTRIDE + k0 + khalf * 16];
        const v8bf hi = *(const v8bf*)&Ab[M * BSTRIDE + k0 + khalf * 16 + 8];
        v16bf a;
#pragma unroll
        for (int e = 0; e < 8; ++e) { a[e] = lo[e]; a[e + 8] = hi[e]; }

        // B fragments: lane holds column (n0+nsel), K = k0+kb .. +15 contiguous
        const v16bf b0 = *(const v16bf*)(wt + (long long)(n0 + nsel) * IN_FEATS + k0 + kb);
        const v16bf b1 = *(const v16bf*)(wt + (long long)(n0 + 16 + nsel) * IN_FEATS + k0 + kb);

        acc0 = __builtin_amdgcn_wmma_f32_16x16x32_bf16(false, a, false, b0, (short)0, acc0, false, false);
        acc1 = __builtin_amdgcn_wmma_f32_16x16x32_bf16(false, a, false, b1, (short)0, acc1, false, false);
    }

    // D layout: VGPR v -> M = v + 8*(lane/16), N = lane%16
    const int dM = (lane >> 4) * 8;
    const int dN = lane & 15;
    if (row0 + 16 <= nNodes) {
        // fast path: no per-row bounds checks (N % 16 == 0 case)
        float* o = ft + (long long)(row0 + dM) * HD + n0 + dN;
#pragma unroll
        for (int v = 0; v < 8; ++v) {
            o[(long long)v * HD]      = acc0[v];
            o[(long long)v * HD + 16] = acc1[v];
        }
    } else {
#pragma unroll
        for (int v = 0; v < 8; ++v) {
            const int r = row0 + dM + v;
            if (r < nNodes) {
                float* o = ft + (long long)r * HD;
                o[n0 + dN]      = acc0[v];
                o[n0 + 16 + dN] = acc1[v];
            }
        }
    }
}

// ---- per-(node,head) attention logits: el = ft.attn_l, er = ft.attn_r ----
__global__ void __launch_bounds__(256) elr_kernel(const float* __restrict__ ft,
                                                  const float* __restrict__ al,
                                                  const float* __restrict__ ar,
                                                  float* __restrict__ el,
                                                  float* __restrict__ er, int nNodes) {
    const int gw   = blockIdx.x * 8 + (threadIdx.x >> 5);   // (node,head) per wave
    const int lane = threadIdx.x & 31;
    if (gw >= nNodes * NUM_HEADS) return;
    const int n = gw >> 2, h = gw & 3;

    const float* f = ft + (long long)n * HD + h * OUT_FEATS;
    const float x0 = f[lane], x1 = f[lane + 32];
    float l = x0 * al[h * 64 + lane] + x1 * al[h * 64 + lane + 32];
    float r = x0 * ar[h * 64 + lane] + x1 * ar[h * 64 + lane + 32];
#pragma unroll
    for (int off = 16; off > 0; off >>= 1) {
        l += __shfl_xor(l, off, 32);
        r += __shfl_xor(r, off, 32);
    }
    if (lane == 0) { el[gw] = l; er[gw] = r; }
}

// ---- pass 1 over edges: e = leaky_relu(el[src]+er[dst]); segment max ----
__global__ void edge_max_kernel(const int* __restrict__ src, const int* __restrict__ dst,
                                const float* __restrict__ el, const float* __restrict__ er,
                                float* __restrict__ eb, float* __restrict__ emax, int nEdges) {
    const int i = blockIdx.x * blockDim.x + threadIdx.x;
    if (i >= nEdges) return;
    const int s = src[i], d = dst[i];
    const float4 l = *(const float4*)(el + s * 4);
    const float4 r = *(const float4*)(er + d * 4);
    float e0 = l.x + r.x; e0 = e0 > 0.0f ? e0 : GAT_ALPHA * e0;
    float e1 = l.y + r.y; e1 = e1 > 0.0f ? e1 : GAT_ALPHA * e1;
    float e2 = l.z + r.z; e2 = e2 > 0.0f ? e2 : GAT_ALPHA * e2;
    float e3 = l.w + r.w; e3 = e3 > 0.0f ? e3 : GAT_ALPHA * e3;
    float4 e; e.x = e0; e.y = e1; e.z = e2; e.w = e3;
    *(float4*)(eb + (long long)i * 4) = e;
    float* mp = emax + d * 4;
    atomicMaxF(mp + 0, e0);
    atomicMaxF(mp + 1, e1);
    atomicMaxF(mp + 2, e2);
    atomicMaxF(mp + 3, e3);
}

// ---- pass 2 over edges: e_exp = exp(e - max[dst]); segment sum ----
__global__ void edge_exp_kernel(const int* __restrict__ dst, float* __restrict__ eb,
                                const float* __restrict__ emax, float* __restrict__ esum,
                                int nEdges) {
    const int i = blockIdx.x * blockDim.x + threadIdx.x;
    if (i >= nEdges) return;
    const int d = dst[i];
    float4 e = *(float4*)(eb + (long long)i * 4);
    const float4 m = *(const float4*)(emax + d * 4);
    e.x = __expf(e.x - m.x);
    e.y = __expf(e.y - m.y);
    e.z = __expf(e.z - m.z);
    e.w = __expf(e.w - m.w);
    *(float4*)(eb + (long long)i * 4) = e;
    float* sp = esum + d * 4;
    atomicAdd(sp + 0, e.x);
    atomicAdd(sp + 1, e.y);
    atomicAdd(sp + 2, e.z);
    atomicAdd(sp + 3, e.w);
}

// ---- pass 3: wave per edge, scatter-add a*ft[src] into out[dst] ----
__global__ void __launch_bounds__(256) scatter_kernel(const int* __restrict__ src,
                                                      const int* __restrict__ dst,
                                                      const float* __restrict__ eb,
                                                      const float* __restrict__ esum,
                                                      const float* __restrict__ ft,
                                                      float* __restrict__ out, int nEdges) {
    const int w = blockIdx.x * 8 + (threadIdx.x >> 5);
    if (w >= nEdges) return;
    const int lane = threadIdx.x & 31;
    const int s = src[w], d = dst[w];
    const float4 e = *(const float4*)(eb + (long long)w * 4);
    const float4 z = *(const float4*)(esum + d * 4);
    const float a0 = e.x / z.x, a1 = e.y / z.y, a2 = e.z / z.z, a3 = e.w / z.w;
    const float* f = ft + (long long)s * HD;
    float* o = out + (long long)d * HD;
#pragma unroll
    for (int k = 0; k < 8; ++k) {
        const int idx = lane + (k << 5);                 // head = k/2 is uniform per k
        const float aa = (k < 2) ? a0 : (k < 4) ? a1 : (k < 6) ? a2 : a3;
        atomicAdd(o + idx, f[idx] * aa);
    }
}

extern "C" void kernel_launch(void* const* d_in, const int* in_sizes, int n_in,
                              void* d_out, int out_size, void* d_ws, size_t ws_size,
                              hipStream_t stream) {
    const float* feat   = (const float*)d_in[0];
    const float* fc_w   = (const float*)d_in[1];
    const float* attn_l = (const float*)d_in[2];
    const float* attn_r = (const float*)d_in[3];
    const int*   src    = (const int*)d_in[4];
    const int*   dst    = (const int*)d_in[5];
    float*       out    = (float*)d_out;

    const int nNodes = in_sizes[0] / IN_FEATS;
    const int nEdges = in_sizes[4];

    // workspace layout (all 16B-aligned)
    char*   ws   = (char*)d_ws;
    size_t  off  = 0;
    __bf16* wt   = (__bf16*)(ws + off); off += (size_t)HD * IN_FEATS * sizeof(__bf16); // 128 KB
    float*  ft   = (float*)(ws + off);  off += (size_t)nNodes * HD * sizeof(float);    // 51.2 MB
    float*  el   = (float*)(ws + off);  off += (size_t)nNodes * NUM_HEADS * sizeof(float);
    float*  er   = (float*)(ws + off);  off += (size_t)nNodes * NUM_HEADS * sizeof(float);
    float*  emax = (float*)(ws + off);  off += (size_t)nNodes * NUM_HEADS * sizeof(float);
    float*  esum = (float*)(ws + off);  off += (size_t)nNodes * NUM_HEADS * sizeof(float);
    float*  eb   = (float*)(ws + off);  off += (size_t)nEdges * NUM_HEADS * sizeof(float);

    // init accumulators / output
    fill_kernel<<<2048, 256, 0, stream>>>(out, (long long)nNodes * HD, 0.0f);
    fill_kernel<<<256, 256, 0, stream>>>(esum, (long long)nNodes * NUM_HEADS, 0.0f);
    fill_kernel<<<256, 256, 0, stream>>>(emax, (long long)nNodes * NUM_HEADS, -__builtin_huge_valf());

    // weight convert/transpose -> bf16 [HD][IN_FEATS]
    wconv_kernel<<<(HD * IN_FEATS) / 256, 256, 0, stream>>>(fc_w, wt);

    // WMMA projection
    gemm_kernel<<<(nNodes + 15) / 16, 256, 0, stream>>>(feat, wt, ft, nNodes);

    // attention logits
    elr_kernel<<<(nNodes * NUM_HEADS + 7) / 8, 256, 0, stream>>>(ft, attn_l, attn_r, el, er, nNodes);

    // edge softmax
    edge_max_kernel<<<(nEdges + 255) / 256, 256, 0, stream>>>(src, dst, el, er, eb, emax, nEdges);
    edge_exp_kernel<<<(nEdges + 255) / 256, 256, 0, stream>>>(dst, eb, emax, esum, nEdges);

    // weighted scatter aggregation
    scatter_kernel<<<(nEdges + 7) / 8, 256, 0, stream>>>(src, dst, eb, esum, ft, out, nEdges);
}
